// Attention_53008486367456
// MI455X (gfx1250) — compile-verified, matrix operand order checked by reading, not run on hardware
//
#include <hip/hip_runtime.h>
#include <hip/hip_bf16.h>
#include <math.h>

typedef __bf16 bf16_t;
typedef __attribute__((ext_vector_type(16))) __bf16 v16bf;
typedef __attribute__((ext_vector_type(8)))  __bf16 v8bf;
typedef __attribute__((ext_vector_type(8)))  float  v8f;

union Frag { v16bf v; v8bf h[2]; };

#define KMASK (-3.3895314e38f)
#define NHEADS 28
#define NKV    4
#define NREP   7
#define HDIM   128

static __device__ __forceinline__ v8f wmma_bf16(const Frag& a, const Frag& b, v8f c) {
    return __builtin_amdgcn_wmma_f32_16x16x32_bf16(false, a.v, false, b.v, (short)0, c, false, false);
}

// async copy of 64 contiguous bytes global -> LDS (ASYNCcnt-tracked, no VGPR roundtrip)
static __device__ __forceinline__ void async_copy64(unsigned lds_addr, const void* gptr) {
    unsigned long long ga = (unsigned long long)(uintptr_t)gptr;
    asm volatile("global_load_async_to_lds_b128 %0, %1, off"           :: "v"(lds_addr), "v"(ga) : "memory");
    asm volatile("global_load_async_to_lds_b128 %0, %1, off offset:16" :: "v"(lds_addr), "v"(ga) : "memory");
    asm volatile("global_load_async_to_lds_b128 %0, %1, off offset:32" :: "v"(lds_addr), "v"(ga) : "memory");
    asm volatile("global_load_async_to_lds_b128 %0, %1, off offset:48" :: "v"(lds_addr), "v"(ga) : "memory");
}
static __device__ __forceinline__ void wait_async() {
    asm volatile("s_wait_asynccnt 0x0" ::: "memory");
}

// ---------------------------------------------------------------------------
// position ids + left pads from segment_ids
// ---------------------------------------------------------------------------
__global__ void posids_kernel(const int* __restrict__ seg, int* __restrict__ pos,
                              int* __restrict__ lp, int B, int T) {
    int b = blockIdx.x;
    int t = threadIdx.x;
    if (t >= T) return;
    int c = 0;
    for (int i = 0; i <= t; ++i) c += (seg[b * T + i] != 0) ? 1 : 0;
    pos[b * T + t] = c - 1;
    if (t == 0) {
        int n = 0;
        for (int i = 0; i < T; ++i) { if (seg[b * T + i] != 0) break; ++n; }
        lp[b] = n;
    }
}

// ---------------------------------------------------------------------------
// flat f32 -> bf16 convert (pre-pass; n divisible by 4)
// ---------------------------------------------------------------------------
__global__ __launch_bounds__(256)
void convert_bf16(const float* __restrict__ in, bf16_t* __restrict__ out, long long n) {
    long long i = ((long long)blockIdx.x * blockDim.x + threadIdx.x) * 4;
    if (i + 3 < n) {
        float4 f = *(const float4*)(in + i);
        out[i]     = (bf16_t)f.x;
        out[i + 1] = (bf16_t)f.y;
        out[i + 2] = (bf16_t)f.z;
        out[i + 3] = (bf16_t)f.w;
    }
}

// ---------------------------------------------------------------------------
// f32 (K x N) -> bf16 transposed (N x K), tiled through LDS (pre-pass)
// ---------------------------------------------------------------------------
__global__ __launch_bounds__(256)
void transpose_convert(const float* __restrict__ in, bf16_t* __restrict__ out,
                       int K, int N) {
    __shared__ float tile[32][33];
    const int k0 = blockIdx.x * 32, n0 = blockIdx.y * 32;
    const int tx = threadIdx.x & 31, ty = threadIdx.x >> 5;   // ty: 0..7
    #pragma unroll
    for (int r = ty; r < 32; r += 8)
        tile[r][tx] = in[(size_t)(k0 + r) * N + n0 + tx];
    __syncthreads();
    #pragma unroll
    for (int r = ty; r < 32; r += 8)
        out[(size_t)(n0 + r) * K + k0 + tx] = (bf16_t)tile[tx][r];
}

// ---------------------------------------------------------------------------
// V transpose: (B,T,NKV,HDIM) f32 -> (B,NKV,HDIM,T) bf16
// ---------------------------------------------------------------------------
__global__ __launch_bounds__(256)
void transpose_v(const float* __restrict__ v, bf16_t* __restrict__ vt, int B, int T) {
    __shared__ float tile[32][33];
    const int t0 = blockIdx.x * 32;
    const int d0 = blockIdx.y * 32;
    const int bk = blockIdx.z;               // b*NKV + kv
    const int b = bk / NKV, kv = bk % NKV;
    const int tx = threadIdx.x & 31, ty = threadIdx.x >> 5;
    #pragma unroll
    for (int r = ty; r < 32; r += 8)
        tile[r][tx] = v[((size_t)(b * T + t0 + r) * NKV + kv) * HDIM + d0 + tx];
    __syncthreads();
    #pragma unroll
    for (int r = ty; r < 32; r += 8)
        vt[((size_t)bk * HDIM + d0 + r) * T + t0 + tx] = (bf16_t)tile[tx][r];
}

// ---------------------------------------------------------------------------
// bf16 WMMA GEMM:  C[MxN](f32) = A[MxK](bf16) * Bt[NxK](bf16)^T  (+ bias[N])
// Block: 256 threads = 8 waves (4x2); block tile 128x128, BK=64.
// Double-buffered async global->LDS pipeline: prefetch tile i+1 while
// computing tile i; one barrier per K-step; 16 v_wmma per step per wave.
// ---------------------------------------------------------------------------
__global__ __launch_bounds__(256)
void gemm_bf16_wmma(const bf16_t* __restrict__ A, const bf16_t* __restrict__ Bt,
                    const float* __restrict__ bias, float* __restrict__ C,
                    int M, int N, int K) {
    __shared__ bf16_t As[2][128][64];   // [buf][m][k]
    __shared__ bf16_t Bs[2][128][64];   // [buf][n][k]
    const int tid  = threadIdx.x;
    const int wave = tid >> 5;
    const int lane = tid & 31;
    const int wm   = wave & 3;       // 32-row strip
    const int wn   = wave >> 2;      // 64-col strip
    const int m    = lane & 15;
    const int hi   = lane >> 4;
    const int m0 = blockIdx.x * 128;
    const int n0 = blockIdx.y * 128;

    const int row = tid >> 1;
    const int cs  = (tid & 1) * 32;
    const bf16_t* ga = A  + (size_t)(m0 + row) * K + cs;
    const bf16_t* gb = Bt + (size_t)(n0 + row) * K + cs;
    const unsigned laA[2] = { (unsigned)(uintptr_t)&As[0][row][cs],
                              (unsigned)(uintptr_t)&As[1][row][cs] };
    const unsigned laB[2] = { (unsigned)(uintptr_t)&Bs[0][row][cs],
                              (unsigned)(uintptr_t)&Bs[1][row][cs] };

    const int ntiles = K >> 6;
    // preload tile 0 into buffer 0
    async_copy64(laA[0], ga);
    async_copy64(laB[0], gb);

    v8f acc[2][4] = {};
    for (int it = 0; it < ntiles; ++it) {
        wait_async();           // my async writes for tile `it` done
        __syncthreads();        // all waves: tile `it` visible; prev buffer free
        const int nxt = it + 1;
        if (nxt < ntiles) {     // prefetch tile it+1 while computing tile it
            const int pb = nxt & 1;
            async_copy64(laA[pb], ga + (size_t)nxt * 64);
            async_copy64(laB[pb], gb + (size_t)nxt * 64);
        }
        const int cur = it & 1;
        #pragma unroll
        for (int kk = 0; kk < 2; ++kk) {
            const int kb0 = kk * 32 + 8 * hi, kb1 = kk * 32 + 16 + 8 * hi;
            Frag af[2], bfr[4];
            #pragma unroll
            for (int mi = 0; mi < 2; ++mi) {
                const int r = wm * 32 + mi * 16 + m;
                af[mi].h[0] = *(const v8bf*)&As[cur][r][kb0];
                af[mi].h[1] = *(const v8bf*)&As[cur][r][kb1];
            }
            #pragma unroll
            for (int ni = 0; ni < 4; ++ni) {
                const int r = wn * 64 + ni * 16 + m;
                bfr[ni].h[0] = *(const v8bf*)&Bs[cur][r][kb0];
                bfr[ni].h[1] = *(const v8bf*)&Bs[cur][r][kb1];
            }
            #pragma unroll
            for (int mi = 0; mi < 2; ++mi)
                #pragma unroll
                for (int ni = 0; ni < 4; ++ni)
                    acc[mi][ni] = wmma_bf16(af[mi], bfr[ni], acc[mi][ni]);
        }
    }

    #pragma unroll
    for (int mi = 0; mi < 2; ++mi) {
        #pragma unroll
        for (int ni = 0; ni < 4; ++ni) {
            const int n = n0 + wn * 64 + ni * 16 + m;
            const float bv = bias ? bias[n] : 0.0f;
            float* cp = C + (size_t)(m0 + wm * 32 + mi * 16 + 8 * hi) * N + n;
            #pragma unroll
            for (int j = 0; j < 8; ++j) cp[(size_t)j * N] = acc[mi][ni][j] + bv;
        }
    }
}

// ---------------------------------------------------------------------------
// RoPE: read f32 Q/K projections, write bf16 roped Q/K
// ---------------------------------------------------------------------------
__global__ __launch_bounds__(256)
void rope_bf16_kernel(const float* __restrict__ q, const float* __restrict__ k,
                      bf16_t* __restrict__ qo, bf16_t* __restrict__ ko,
                      const int* __restrict__ pos, int B, int T) {
    const long long Nq = (long long)B * T * NHEADS * 64;
    const long long Nk = (long long)B * T * NKV * 64;
    long long idx = (long long)blockIdx.x * blockDim.x + threadIdx.x;
    const float* base; bf16_t* obase; int heads; long long id;
    if (idx < Nq)           { base = q; obase = qo; heads = NHEADS; id = idx; }
    else if (idx < Nq + Nk) { base = k; obase = ko; heads = NKV;    id = idx - Nq; }
    else return;
    int i = (int)(id % 64); id /= 64;
    int h = (int)(id % heads); id /= heads;
    int t = (int)(id % T);
    int b = (int)(id / T);
    int p = pos[b * T + t];
    // theta^(-i/64) = exp2(-log2(theta) * i/64)
    float inv = exp2f(-19.931568569324174f * ((float)i * (1.0f / 64.0f)));
    float ang = (float)p * inv;
    float s = __sinf(ang), c = __cosf(ang);
    const float* xp = base + ((size_t)(b * T + t) * heads + h) * HDIM;
    bf16_t* op = obase + ((size_t)(b * T + t) * heads + h) * HDIM;
    float x1 = xp[i], x2 = xp[i + 64];
    op[i]      = (bf16_t)(x1 * c - x2 * s);
    op[i + 64] = (bf16_t)(x2 * c + x1 * s);
}

// ---------------------------------------------------------------------------
// Flash attention (bf16 in, bf16 out): block = 4 waves, wave = 16-query tile
// V is pre-transposed (B,NKV,HDIM,T) so all fragments are vector loads.
// ---------------------------------------------------------------------------
__global__ __launch_bounds__(128)
void attn_kernel(const bf16_t* __restrict__ Q,   // (B,T,28,128) roped bf16
                 const bf16_t* __restrict__ Kc,  // (B,T,4,128)  roped bf16
                 const bf16_t* __restrict__ Vt,  // (B,4,128,T)  bf16
                 const int* __restrict__ lp, const int* __restrict__ seg,
                 bf16_t* __restrict__ O,         // (B,T,28,128) bf16
                 int B, int T) {
    __shared__ bf16_t Pl[4][16][32];
    const int wave = threadIdx.x >> 5;
    const int lane = threadIdx.x & 31;
    const int m    = lane & 15;
    const int hi   = lane >> 4;

    int gid = blockIdx.x;
    const int grp = T / 64;
    const int qt4 = gid % grp; gid /= grp;
    const int h   = gid % NHEADS;
    const int b   = gid / NHEADS;
    const int q0  = (qt4 * 4 + wave) * 16;
    const int kv  = h / NREP;

    // Q fragments: two 16B vector loads each
    Frag qf[4];
    const bf16_t* qrow = Q + ((size_t)(b * T + q0 + m) * NHEADS + h) * HDIM;
    #pragma unroll
    for (int kk = 0; kk < 4; ++kk) {
        qf[kk].h[0] = *(const v8bf*)(qrow + kk * 32 + 8 * hi);
        qf[kk].h[1] = *(const v8bf*)(qrow + kk * 32 + 16 + 8 * hi);
    }

    int qidx[8], segq[8];
    #pragma unroll
    for (int j = 0; j < 8; ++j) {
        qidx[j] = q0 + j + 8 * hi;
        segq[j] = seg[b * T + qidx[j]];
    }
    const int lpb = lp[b];

    float rm[8], rl[8];
    #pragma unroll
    for (int j = 0; j < 8; ++j) { rm[j] = KMASK; rl[j] = 0.0f; }
    v8f acc[8] = {};

    const float scale = 0.088388347648318447f;  // 128^-0.5
    const int smax = (q0 + 16 < T) ? q0 + 16 : T;
    const bf16_t* vbase = Vt + ((size_t)(b * NKV + kv) * HDIM + m) * T;

    for (int s0 = 0; s0 < smax; s0 += 32) {
        // ---- scores ----
        v8f sc[2] = {};
        #pragma unroll
        for (int c = 0; c < 2; ++c) {
            const bf16_t* krow = Kc + ((size_t)(b * T + s0 + c * 16 + m) * NKV + kv) * HDIM;
            #pragma unroll
            for (int kk = 0; kk < 4; ++kk) {
                Frag bfr;
                bfr.h[0] = *(const v8bf*)(krow + kk * 32 + 8 * hi);
                bfr.h[1] = *(const v8bf*)(krow + kk * 32 + 16 + 8 * hi);
                sc[c] = wmma_bf16(qf[kk], bfr, sc[c]);
            }
        }
        // ---- scale + mask ----
        #pragma unroll
        for (int c = 0; c < 2; ++c) {
            const int s_col = s0 + c * 16 + m;
            const bool kvseg = (s_col >= lpb) && (s_col < T);
            #pragma unroll
            for (int j = 0; j < 8; ++j) {
                const bool ok = (s_col <= qidx[j]) && ((kvseg ? 1 : 0) == segq[j]);
                sc[c][j] = ok ? sc[c][j] * scale : KMASK;
            }
        }
        // ---- online softmax ----
        float rmax[8], rsum[8], alpha[8];
        #pragma unroll
        for (int j = 0; j < 8; ++j) rmax[j] = fmaxf(sc[0][j], sc[1][j]);
        #pragma unroll
        for (int off = 1; off < 16; off <<= 1)
            #pragma unroll
            for (int j = 0; j < 8; ++j) rmax[j] = fmaxf(rmax[j], __shfl_xor(rmax[j], off, 32));
        #pragma unroll
        for (int j = 0; j < 8; ++j) {
            float mn = fmaxf(rm[j], rmax[j]);
            alpha[j] = __expf(rm[j] - mn);
            rm[j] = mn;
            float p0 = __expf(sc[0][j] - mn);
            float p1 = __expf(sc[1][j] - mn);
            sc[0][j] = p0; sc[1][j] = p1;
            rsum[j] = p0 + p1;
        }
        #pragma unroll
        for (int off = 1; off < 16; off <<= 1)
            #pragma unroll
            for (int j = 0; j < 8; ++j) rsum[j] += __shfl_xor(rsum[j], off, 32);
        #pragma unroll
        for (int j = 0; j < 8; ++j) rl[j] = rl[j] * alpha[j] + rsum[j];
        #pragma unroll
        for (int dn = 0; dn < 8; ++dn)
            #pragma unroll
            for (int j = 0; j < 8; ++j) acc[dn][j] *= alpha[j];

        // ---- stage P through LDS (C layout -> A fragment), wave-local ----
        #pragma unroll
        for (int c = 0; c < 2; ++c)
            #pragma unroll
            for (int j = 0; j < 8; ++j)
                Pl[wave][j + 8 * hi][c * 16 + m] = (bf16_t)sc[c][j];
        Frag pf;   // DS ops within a wave are in-order: RAW through LDS is safe
        pf.h[0] = *(const v8bf*)&Pl[wave][m][8 * hi];
        pf.h[1] = *(const v8bf*)&Pl[wave][m][16 + 8 * hi];

        // ---- acc += P * V : V fragments are contiguous vector loads ----
        #pragma unroll
        for (int dn = 0; dn < 8; ++dn) {
            const bf16_t* vrow = vbase + (size_t)(dn * 16) * T;   // d = dn*16 + m
            Frag vf;
            vf.h[0] = *(const v8bf*)(vrow + s0 + 8 * hi);
            vf.h[1] = *(const v8bf*)(vrow + s0 + 16 + 8 * hi);
            acc[dn] = wmma_bf16(pf, vf, acc[dn]);
        }
    }

    #pragma unroll
    for (int dn = 0; dn < 8; ++dn) {
        const int d = dn * 16 + m;
        #pragma unroll
        for (int j = 0; j < 8; ++j) {
            O[((size_t)(b * T + qidx[j]) * NHEADS + h) * HDIM + d] =
                (bf16_t)(acc[dn][j] / rl[j]);
        }
    }
}

// ---------------------------------------------------------------------------
extern "C" void kernel_launch(void* const* d_in, const int* in_sizes, int n_in,
                              void* d_out, int out_size, void* d_ws, size_t ws_size,
                              hipStream_t stream) {
    const int D = 3584, KVD = 512;
    const float* x   = (const float*)d_in[0];
    const int*   seg = (const int*)d_in[1];
    const float* Wq  = (const float*)d_in[4];
    const float* bq  = (const float*)d_in[5];
    const float* Wk  = (const float*)d_in[6];
    const float* bk  = (const float*)d_in[7];
    const float* Wv  = (const float*)d_in[8];
    const float* bv  = (const float*)d_in[9];
    const float* Wo  = (const float*)d_in[10];
    float* out = (float*)d_out;

    const int BT = in_sizes[1];      // B*T
    const int B  = 2;
    const int T  = BT / B;
    const size_t BTs = (size_t)BT;

    float*  qbuf = (float*)d_ws;                        // BT*D  f32
    float*  kbuf = qbuf + BTs * D;                      // BT*KVD f32
    float*  vbuf = kbuf + BTs * KVD;                    // BT*KVD f32
    bf16_t* xbf  = (bf16_t*)(vbuf + BTs * KVD);         // BT*D
    bf16_t* WqT  = xbf + BTs * D;                       // D*D
    bf16_t* WkT  = WqT + (size_t)D * D;                 // KVD*D
    bf16_t* WvT  = WkT + (size_t)KVD * D;               // KVD*D
    bf16_t* WoT  = WvT + (size_t)KVD * D;               // D*D
    bf16_t* qbf  = WoT + (size_t)D * D;                 // BT*D
    bf16_t* kbf  = qbf + BTs * D;                       // BT*KVD
    bf16_t* vtbf = kbf + BTs * KVD;                     // BT*KVD (transposed V)
    int*    pos  = (int*)(vtbf + BTs * KVD);            // BT
    int*    lpb  = pos + BT;                            // B
    bf16_t* obf  = (bf16_t*)qbuf;                       // reuse (qbuf dead after rope)

    posids_kernel<<<B, T, 0, stream>>>(seg, pos, lpb, B, T);

    // pre-pass conversions / transposes (one-shot, amortized)
    convert_bf16<<<(int)((BTs * D / 4 + 255) / 256), 256, 0, stream>>>(x, xbf, BTs * D);
    transpose_convert<<<dim3(D / 32, D / 32), 256, 0, stream>>>(Wq, WqT, D, D);
    transpose_convert<<<dim3(D / 32, KVD / 32), 256, 0, stream>>>(Wk, WkT, D, KVD);
    transpose_convert<<<dim3(D / 32, KVD / 32), 256, 0, stream>>>(Wv, WvT, D, KVD);
    transpose_convert<<<dim3(D / 32, D / 32), 256, 0, stream>>>(Wo, WoT, D, D);

    // QKV projections
    gemm_bf16_wmma<<<dim3(BT / 128, D / 128), 256, 0, stream>>>(xbf, WqT, bq, qbuf, BT, D, D);
    gemm_bf16_wmma<<<dim3(BT / 128, KVD / 128), 256, 0, stream>>>(xbf, WkT, bk, kbuf, BT, KVD, D);
    gemm_bf16_wmma<<<dim3(BT / 128, KVD / 128), 256, 0, stream>>>(xbf, WvT, bv, vbuf, BT, KVD, D);

    // RoPE -> bf16 Q/K; V -> transposed bf16
    const long long rope_total = (long long)BT * (NHEADS + NKV) * 64;
    rope_bf16_kernel<<<(int)((rope_total + 255) / 256), 256, 0, stream>>>(qbuf, kbuf, qbf, kbf, pos, B, T);
    transpose_v<<<dim3(T / 32, HDIM / 32, B * NKV), 256, 0, stream>>>(vbuf, vtbf, B, T);

    // attention
    attn_kernel<<<B * NHEADS * (T / 64), 128, 0, stream>>>(qbf, kbf, vtbf, lpb, seg, obf, B, T);

    // output projection
    gemm_bf16_wmma<<<dim3(BT / 128, D / 128), 256, 0, stream>>>(obf, WoT, nullptr, out, BT, D, D);
}
